// MultiHeadGraphAttention_33011118637078
// MI455X (gfx1250) — compile-verified, matrix operand order checked by reading, not run on hardware
//
#include <hip/hip_runtime.h>
#include <hip/hip_bf16.h>

#define N_NODES 8192
#define IN_F    256
#define OUT_F   256
#define N_HEADS 4
#define DH      64
#define ALPHA   0.2f
#define NEGV    (-9.0e15f)

// attn kernel tiling
#define ROWS 16          // rows per block
#define JC   128         // adjacency columns staged per async chunk
#define RS   132         // padded LDS row stride in ints -> conflict-free ds_load_b128
#define NCHUNK (N_NODES / JC)

typedef __attribute__((ext_vector_type(16))) __bf16 v16bf;
typedef __attribute__((ext_vector_type(8)))  __bf16 v8bf;
typedef __attribute__((ext_vector_type(2)))  __bf16 v2bf;
typedef __attribute__((ext_vector_type(8)))  float  v8f;

union V16U { v16bf v; v8bf h[2]; v2bf p[8]; };

static __device__ __forceinline__ __bf16 f2bf(float f) {
    unsigned u = __builtin_bit_cast(unsigned, f);
    unsigned r = (u + 0x7FFFu + ((u >> 16) & 1u)) >> 16;  // round-to-nearest-even
    return __builtin_bit_cast(__bf16, (unsigned short)r);
}

// packed f32x2 -> bf16x2: 2x v_add (round-half-up) + 1x v_perm_b32
static __device__ __forceinline__ v2bf f2bf_pk(float lo, float hi) {
    unsigned ul = __builtin_bit_cast(unsigned, lo) + 0x8000u;
    unsigned uh = __builtin_bit_cast(unsigned, hi) + 0x8000u;
    unsigned pk = __builtin_amdgcn_perm(uh, ul, 0x07060302u);  // {uh[31:16], ul[31:16]}
    return __builtin_bit_cast(v2bf, pk);
}

static __device__ __forceinline__ void wait_async0() {
#if __has_builtin(__builtin_amdgcn_s_wait_asynccnt)
    __builtin_amdgcn_s_wait_asynccnt(0);
#else
    asm volatile("s_wait_asynccnt 0x0" ::: "memory");
#endif
}

// ---------------------------------------------------------------- K1: f32 -> bf16
__global__ void gat_cvt_bf16(const float* __restrict__ src, __bf16* __restrict__ dst, int n) {
    int i = blockIdx.x * blockDim.x + threadIdx.x;
    if (i < n) dst[i] = f2bf(src[i]);
}

// ---------------------------------------------------------------- K2: Wh = h @ W  (per head)
// grid: (N/16, H); block 128 = 4 waves; wave w -> d-tile w (DH/16 == 4)
// Outputs: Whf32 [H][N][DH] row-major, WhbfT [H][DH][N] (transposed, bf16)
__global__ void gat_wh_kernel(const float* __restrict__ h, const float* __restrict__ W,
                              float* __restrict__ Whf32, __bf16* __restrict__ WhbfT) {
    const int wave = threadIdx.x >> 5;
    const int lane = threadIdx.x & 31;
    const int i0   = blockIdx.x * 16;
    const int hh   = blockIdx.y;
    const int d0   = wave * 16;
    const int m    = lane & 15;
    const int off  = (lane < 16) ? 0 : 8;   // K-half per ISA A/B layout
    const int arow = i0 + m;                // A-operand row
    const int bcol = d0 + m;                // B-operand column

    v8f acc = {};
    for (int kc = 0; kc < IN_F; kc += 32) {
        const float* hp = h + (size_t)arow * IN_F + kc + off;
        V16U a, b;
#pragma unroll
        for (int t = 0; t < 4; ++t) {
            a.p[t]     = f2bf_pk(hp[2 * t],      hp[2 * t + 1]);
            a.p[4 + t] = f2bf_pk(hp[16 + 2 * t], hp[16 + 2 * t + 1]);
        }
        const float* wp = W + ((size_t)hh * IN_F) * DH + bcol;  // W[h][k][d]
#pragma unroll
        for (int t = 0; t < 4; ++t) {
            b.p[t]     = f2bf_pk(wp[(size_t)(kc + off + 2 * t) * DH],
                                 wp[(size_t)(kc + off + 2 * t + 1) * DH]);
            b.p[4 + t] = f2bf_pk(wp[(size_t)(kc + 16 + off + 2 * t) * DH],
                                 wp[(size_t)(kc + 16 + off + 2 * t + 1) * DH]);
        }
        acc = __builtin_amdgcn_wmma_f32_16x16x32_bf16(false, a.v, false, b.v, (short)0, acc, false, false);
    }
    const int rb = (lane < 16) ? 0 : 8;
    const int gd = d0 + (lane & 15);
#pragma unroll
    for (int r = 0; r < 8; r += 2) {
        const int gi = i0 + rb + r;
        Whf32[(((size_t)hh * N_NODES) + gi)     * DH + gd] = acc[r];
        Whf32[(((size_t)hh * N_NODES) + gi + 1) * DH + gd] = acc[r + 1];
        // consecutive rows gi, gi+1 are adjacent in transposed layout -> one b32 store
        *(v2bf*)(WhbfT + (((size_t)hh * DH) + gd) * N_NODES + gi) = f2bf_pk(acc[r], acc[r + 1]);
    }
}

// ---------------------------------------------------------------- K3: Wh1/Wh2 dot products
__global__ void gat_wh12_kernel(const float* __restrict__ Whf32,
                                const float* __restrict__ a1, const float* __restrict__ a2,
                                float* __restrict__ Wh1, float* __restrict__ Wh2) {
    int idx = blockIdx.x * blockDim.x + threadIdx.x;  // [H][N] flat
    if (idx >= N_HEADS * N_NODES) return;
    const int hh = idx / N_NODES;
    const float* wr = Whf32 + (size_t)idx * DH;
    const float* p1 = a1 + hh * DH;
    const float* p2 = a2 + hh * DH;
    float s1 = 0.f, s2 = 0.f;
#pragma unroll 8
    for (int d = 0; d < DH; ++d) { float v = wr[d]; s1 += v * p1[d]; s2 += v * p2[d]; }
    Wh1[idx] = s1;
    Wh2[idx] = s2;
}

// ---------------------------------------------------------------- K4: fused masked softmax + PV
// grid: N/16 blocks; block 128 = 4 waves; wave = head.
// Adjacency (268 MB stream) is double-buffered through LDS via async global->LDS loads
// (GLOBAL_LOAD_ASYNC_TO_LDS_B128, ASYNCcnt), decoupled from the WMMA/exp pipeline.
// Dynamic LDS: 2 buffers of 16 rows x 132-int padded stride = 16896 bytes, at LDS offset 0.
__global__ void gat_attn_kernel(const int* __restrict__ adj, const __bf16* __restrict__ WhbfT,
                                const float* __restrict__ Wh1, const float* __restrict__ Wh2,
                                __bf16* __restrict__ concatbf) {
    extern __shared__ int adjbuf[];        // [2][ROWS*RS], starts at LDS byte offset 0

    const int hh   = threadIdx.x >> 5;     // head
    const int lane = threadIdx.x & 31;
    const int i0   = blockIdx.x * ROWS;
    const int m    = lane & 15;
    const int off  = (lane < 16) ? 0 : 8;
    const int gi   = i0 + m;

    const float wh1_i = Wh1[(size_t)hh * N_NODES + gi];
    const float* wh2r = Wh2 + (size_t)hh * N_NODES;

    float m_run = -__builtin_inff();
    float l_run = 0.f;
    v8f acc0 = {}, acc1 = {}, acc2 = {}, acc3 = {};

    // cooperative async stage of one 16 x 128 int32 adjacency chunk into LDS buffer (c & 1)
    auto issue_chunk = [&](int c) {
        const int buf = c & 1;
        const int jb  = c * JC;
#pragma unroll
        for (int k = 0; k < 4; ++k) {
            const int s   = (int)threadIdx.x + k * 128;   // 512 x 16B segments
            const int row = s >> 5;                        // 32 segments per row
            const int seg = s & 31;
            const unsigned long long ga =
                (unsigned long long)(adj + (size_t)(i0 + row) * N_NODES + jb + seg * 4);
            const unsigned lofs = (unsigned)(buf * (ROWS * RS * 4) + row * (RS * 4) + seg * 16);
            asm volatile("global_load_async_to_lds_b128 %0, %1, off"
                         :: "v"(lofs), "v"(ga) : "memory");
        }
    };

    issue_chunk(0);

    for (int c = 0; c < NCHUNK; ++c) {
        wait_async0();        // my wave's in-flight async LDS writes complete
        __syncthreads();      // all waves' chunk data visible; prior buffer fully consumed
        if (c + 1 < NCHUNK) issue_chunk(c + 1);

        const int* bufp = adjbuf + (c & 1) * (ROWS * RS) + m * RS;

#pragma unroll
        for (int sub = 0; sub < 4; ++sub) {
            const int j0 = c * JC + sub * 32;
            const float* w2 = wh2r + j0 + off;
            const int*   lp = bufp + sub * 32 + off;      // ds_load_b128-friendly

            float ev[16];
            float cmax = -__builtin_inff();
#pragma unroll
            for (int g = 0; g < 2; ++g) {
                const int ko = g * 16;
#pragma unroll
                for (int t = 0; t < 8; ++t) {
                    float e = wh1_i + w2[ko + t];
                    e = (e > 0.f) ? e : ALPHA * e;            // leaky relu
                    e = (lp[ko + t] > 0) ? e : NEGV;          // adjacency mask (from LDS)
                    ev[g * 8 + t] = e;
                    cmax = fmaxf(cmax, e);
                }
            }
            cmax = fmaxf(cmax, __shfl_xor(cmax, 16, 32));     // combine K-halves (wave32)
            const float m_new = fmaxf(m_run, cmax);
            const float scale = __expf(m_run - m_new);

            float csum = 0.f;
            float pv[16];
#pragma unroll
            for (int t = 0; t < 16; ++t) {
                pv[t] = __expf(ev[t] - m_new);
                csum += pv[t];
            }
            V16U a;
#pragma unroll
            for (int t = 0; t < 8; ++t) a.p[t] = f2bf_pk(pv[2 * t], pv[2 * t + 1]);

            csum += __shfl_xor(csum, 16, 32);
            l_run = l_run * scale + csum;
            m_run = m_new;

            // rescale accumulators: VGPR r holds row (r + 8*(lane>=16)); scale lives in lane (row%16)
            const int sb = (lane < 16) ? 0 : 8;
#pragma unroll
            for (int r = 0; r < 8; ++r) {
                const float s = __shfl(scale, sb + r, 32);
                acc0[r] *= s; acc1[r] *= s; acc2[r] *= s; acc3[r] *= s;
            }

            // B operand: WhbfT[h][d][j] -> contiguous 16B loads along K (4 MB total, L2-resident)
            const __bf16* bt = WhbfT + (size_t)hh * DH * N_NODES + j0 + off;
            const int n = lane & 15;
#pragma unroll
            for (int dt = 0; dt < 4; ++dt) {
                const __bf16* bp = bt + (size_t)(dt * 16 + n) * N_NODES;
                V16U b;
                b.h[0] = *(const v8bf*)(bp);
                b.h[1] = *(const v8bf*)(bp + 16);
                if      (dt == 0) acc0 = __builtin_amdgcn_wmma_f32_16x16x32_bf16(false, a.v, false, b.v, (short)0, acc0, false, false);
                else if (dt == 1) acc1 = __builtin_amdgcn_wmma_f32_16x16x32_bf16(false, a.v, false, b.v, (short)0, acc1, false, false);
                else if (dt == 2) acc2 = __builtin_amdgcn_wmma_f32_16x16x32_bf16(false, a.v, false, b.v, (short)0, acc2, false, false);
                else              acc3 = __builtin_amdgcn_wmma_f32_16x16x32_bf16(false, a.v, false, b.v, (short)0, acc3, false, false);
            }
        }
    }

    // normalize, ELU, write concat (bf16) [N][OUT_F]
    const float rl = 1.0f / l_run;
    const int sb = (lane < 16) ? 0 : 8;
#pragma unroll
    for (int r = 0; r < 8; ++r) {
        const float rn = __shfl(rl, sb + r, 32);
        const int grow = i0 + sb + r;
        __bf16* orow = concatbf + (size_t)grow * OUT_F + hh * DH + (lane & 15);
        float v0 = acc0[r] * rn; v0 = (v0 > 0.f) ? v0 : (__expf(v0) - 1.f);
        float v1 = acc1[r] * rn; v1 = (v1 > 0.f) ? v1 : (__expf(v1) - 1.f);
        float v2 = acc2[r] * rn; v2 = (v2 > 0.f) ? v2 : (__expf(v2) - 1.f);
        float v3 = acc3[r] * rn; v3 = (v3 > 0.f) ? v3 : (__expf(v3) - 1.f);
        v2bf p01 = f2bf_pk(v0, v1);
        v2bf p23 = f2bf_pk(v2, v3);
        orow[0]  = p01[0];
        orow[16] = p01[1];
        orow[32] = p23[0];
        orow[48] = p23[1];
    }
}

// ---------------------------------------------------------------- K5: out = concat @ Wp^T + bp
// grid: (N/16, OUT_F/64); block 128 = 4 waves; wave w -> o-tile
__global__ void gat_proj_kernel(const __bf16* __restrict__ concatbf, const __bf16* __restrict__ Wpbf,
                                const float* __restrict__ bp, float* __restrict__ out) {
    const int wave = threadIdx.x >> 5;
    const int lane = threadIdx.x & 31;
    const int i0 = blockIdx.x * 16;
    const int o0 = (blockIdx.y * 4 + wave) * 16;
    const int m  = lane & 15;
    const int off = (lane < 16) ? 0 : 8;
    const int gi = i0 + m;     // A row
    const int go = o0 + m;     // B column (Wp row, since B[k][n] = Wp[n][k])

    v8f acc = {};
    for (int kc = 0; kc < OUT_F; kc += 32) {
        const __bf16* ap = concatbf + (size_t)gi * OUT_F + kc + off;
        const __bf16* wp = Wpbf     + (size_t)go * OUT_F + kc + off;
        V16U a, b;
        a.h[0] = *(const v8bf*)(ap);
        a.h[1] = *(const v8bf*)(ap + 16);
        b.h[0] = *(const v8bf*)(wp);
        b.h[1] = *(const v8bf*)(wp + 16);
        acc = __builtin_amdgcn_wmma_f32_16x16x32_bf16(false, a.v, false, b.v, (short)0, acc, false, false);
    }
    const int sb = (lane < 16) ? 0 : 8;
#pragma unroll
    for (int r = 0; r < 8; ++r) {
        const int grow = i0 + sb + r;
        const int gcol = o0 + (lane & 15);
        out[(size_t)grow * OUT_F + gcol] = acc[r] + bp[gcol];
    }
}

// ---------------------------------------------------------------- launch
extern "C" void kernel_launch(void* const* d_in, const int* in_sizes, int n_in,
                              void* d_out, int out_size, void* d_ws, size_t ws_size,
                              hipStream_t stream) {
    const float* h   = (const float*)d_in[0];
    const int*   adj = (const int*)  d_in[1];
    const float* W   = (const float*)d_in[2];
    const float* a1  = (const float*)d_in[3];
    const float* a2  = (const float*)d_in[4];
    const float* Wp  = (const float*)d_in[5];
    const float* bp  = (const float*)d_in[6];
    float* out = (float*)d_out;

    char* ws = (char*)d_ws;
    constexpr size_t SZ_WHBFT  = (size_t)N_HEADS * DH * N_NODES * 2;      // 4 MB
    constexpr size_t SZ_WHF32  = (size_t)N_HEADS * N_NODES * DH * 4;      // 8 MB
    constexpr size_t SZ_WH12   = (size_t)N_HEADS * N_NODES * 4;           // 128 KB each
    constexpr size_t SZ_CONCAT = (size_t)N_NODES * OUT_F * 2;             // 4 MB

    __bf16* WhbfT    = (__bf16*)(ws);
    float*  Whf32    = (float*) (ws + SZ_WHBFT);
    float*  Wh1      = (float*) (ws + SZ_WHBFT + SZ_WHF32);
    float*  Wh2      = (float*) (ws + SZ_WHBFT + SZ_WHF32 + SZ_WH12);
    __bf16* concatbf = (__bf16*)(ws + SZ_WHBFT + SZ_WHF32 + 2 * SZ_WH12);
    __bf16* Wpbf     = (__bf16*)(ws + SZ_WHBFT + SZ_WHF32 + 2 * SZ_WH12 + SZ_CONCAT);

    // K1: Wp -> bf16
    gat_cvt_bf16<<<(OUT_F * OUT_F + 255) / 256, 256, 0, stream>>>(Wp, Wpbf, OUT_F * OUT_F);

    // K2: Wh GEMM (WMMA bf16, f32 accumulate)
    dim3 g2(N_NODES / 16, N_HEADS);
    gat_wh_kernel<<<g2, 128, 0, stream>>>(h, W, Whf32, WhbfT);

    // K3: attention logit vectors
    gat_wh12_kernel<<<(N_HEADS * N_NODES + 255) / 256, 256, 0, stream>>>(Whf32, a1, a2, Wh1, Wh2);

    // K4: fused masked softmax + attn@Wh (flash-style, async-LDS adjacency, WMMA PV)
    constexpr size_t LDS_BYTES = 2ull * ROWS * RS * 4;                     // 16896 B
    gat_attn_kernel<<<N_NODES / 16, 128, LDS_BYTES, stream>>>(adj, WhbfT, Wh1, Wh2, concatbf);

    // K5: output projection (WMMA)
    dim3 g5(N_NODES / 16, OUT_F / 64);
    gat_proj_kernel<<<g5, 128, 0, stream>>>(concatbf, Wpbf, bp, out);
}
